// DifferentiableFSA_78245714198986
// MI455X (gfx1250) — matrix operand order repaired and checked by reading, not compiled, not run on hardware
//
#include <hip/hip_runtime.h>
#include <math.h>

#define DEV static __device__ __forceinline__

typedef __attribute__((ext_vector_type(16))) _Float16 v16h;
typedef __attribute__((ext_vector_type(8)))  _Float16 v8h;
typedef __attribute__((ext_vector_type(8)))  float    v8f;

// ---------------------------------------------------------------------------
// WMMA fragment loaders (CDNA5 16x16x32 f16 layouts, wave32)
// A (16x32, MxK): lane l -> row M=l&15, khalf=l>>4.
//   element e: k = ((e&8)<<1) + khalf*8 + (e&7)
// B (32x16, KxN) from "Bt" storage (N x K row-major): lane l -> col N=l&15,
//   element e: k = ((e&8)<<1) + (l>>4)*8 + (e&7)   (2x b128)
// C/D (16x16 f32): vgpr i -> row = i + 8*(l>>4), col = l&15
// ---------------------------------------------------------------------------
DEV v16h load_frag_A(const _Float16* A, int lda, int m0, int k0, int klim, int lane) {
  const int row = m0 + (lane & 15);
  const int kh  = (lane >> 4) << 3;         // 0 or 8
  const _Float16* p = A + row * lda + k0;
  v16h a;
  if (klim >= 32) {
    v8h lo = *(const v8h*)(p + kh);
    v8h hi = *(const v8h*)(p + 16 + kh);
#pragma unroll
    for (int e = 0; e < 8; ++e) { a[e] = lo[e]; a[e + 8] = hi[e]; }
  } else {
#pragma unroll
    for (int e = 0; e < 16; ++e) {
      int lk = ((e & 8) << 1) + kh + (e & 7);
      a[e] = (lk < klim) ? p[lk] : (_Float16)0.0f;
    }
  }
  return a;
}

DEV v16h load_frag_B(const _Float16* Bt, int ldb, int n0, int k0, int klim, int lane) {
  const int col = n0 + (lane & 15);
  const int kh  = (lane >> 4) << 3;         // 0 or 8
  const _Float16* p = Bt + col * ldb + k0;
  v16h b;
  if (klim >= 32) {
    v8h lo = *(const v8h*)(p + kh);
    v8h hi = *(const v8h*)(p + 16 + kh);
#pragma unroll
    for (int e = 0; e < 8; ++e) { b[e] = lo[e]; b[e + 8] = hi[e]; }
  } else {
#pragma unroll
    for (int e = 0; e < 16; ++e) {
      int lk = ((e & 8) << 1) + kh + (e & 7);
      b[e] = (lk < klim) ? p[lk] : (_Float16)0.0f;
    }
  }
  return b;
}

// M = 64 fixed: wave w owns row tile w. Epilogue gets (row, col, acc_value).
template <int N, int K, class Epi>
DEV void wmma_gemm(const _Float16* A, int lda, const _Float16* Bt, int ldb,
                   int tid, Epi epi) {
  const int wave = tid >> 5, lane = tid & 31;
  const int m0 = wave << 4;
  for (int n0 = 0; n0 < N; n0 += 16) {
    v8f acc = {0.f, 0.f, 0.f, 0.f, 0.f, 0.f, 0.f, 0.f};
#pragma unroll
    for (int k0 = 0; k0 < K; k0 += 32) {
      const int klim = (K - k0 < 32) ? (K - k0) : 32;
      v16h a = load_frag_A(A, lda, m0, k0, klim, lane);
      v16h b = load_frag_B(Bt, ldb, n0, k0, klim, lane);
      acc = __builtin_amdgcn_wmma_f32_16x16x32_f16(
          false, a, false, b, (short)0, acc, false, false);
    }
    const int row0 = m0 + ((lane >> 4) << 3);
    const int col  = n0 + (lane & 15);
#pragma unroll
    for (int i = 0; i < 8; ++i) epi(row0 + i, col, acc[i]);
  }
}

// ---------------------------------------------------------------------------
// LayerNorm over H=64, one row per thread (threads 0..63)
// ---------------------------------------------------------------------------
DEV void layernorm16(const float* x, const float* g, const float* b,
                     _Float16* out, int tid) {
  if (tid < 64) {
    const float* r = x + tid * 64;
    float mu = 0.f;
    for (int j = 0; j < 64; ++j) mu += r[j];
    mu *= 0.015625f;
    float var = 0.f;
    for (int j = 0; j < 64; ++j) { float d = r[j] - mu; var += d * d; }
    var *= 0.015625f;
    float rs = rsqrtf(var + 1e-6f);
    _Float16* o = out + tid * 64;
    for (int j = 0; j < 64; ++j)
      o[j] = (_Float16)(g[j] * (r[j] - mu) * rs + b[j]);
  }
}

DEV void layernorm32(const float* x, const float* g, const float* b,
                     float* out, int tid) {
  if (tid < 64) {
    const float* r = x + tid * 64;
    float mu = 0.f;
    for (int j = 0; j < 64; ++j) mu += r[j];
    mu *= 0.015625f;
    float var = 0.f;
    for (int j = 0; j < 64; ++j) { float d = r[j] - mu; var += d * d; }
    var *= 0.015625f;
    float rs = rsqrtf(var + 1e-6f);
    float* o = out + tid * 64;
    for (int j = 0; j < 64; ++j) o[j] = g[j] * (r[j] - mu) * rs + b[j];
  }
}

// differentiable base-10 assembly (serial, thread 0)
DEV float soft_assemble(const float* m, const float* dig) {
  float tot = 0.f;
  for (int i = 0; i < 64; ++i) tot += m[i];
  float cum = 0.f, val = 0.f;
  for (int i = 0; i < 64; ++i) {
    cum += m[i];
    float w = __powf(10.f, (tot - cum) * m[i]) * m[i];
    val += dig[i] * w;
  }
  return val;
}

// ---------------------------------------------------------------------------
// Weight pre-pack: fp32 -> f16, transposed to N x K for B-fragment loads.
// Per layer (40960 halves): wq_t[64x64] wk_t aw{q,k,v,o}_t w1_t[128x64] w2_t[64x128]
// ---------------------------------------------------------------------------
__global__ void convert_weights(const float* __restrict__ gq,
                                const float* __restrict__ gk,
                                const float* __restrict__ aw,
                                const float* __restrict__ f1,
                                const float* __restrict__ f2,
                                _Float16* __restrict__ ws) {
  int idx = blockIdx.x * 256 + threadIdx.x;
  if (idx >= 2 * 40960) return;
  int li = idx / 40960, r = idx % 40960;
  _Float16* dst = ws + li * 40960;
  if (r < 4096) {
    int n = r >> 6, k = r & 63;
    dst[r] = (_Float16)gq[li * 4096 + k * 64 + n];
  } else if (r < 8192) {
    int q = r - 4096; int n = q >> 6, k = q & 63;
    dst[r] = (_Float16)gk[li * 4096 + k * 64 + n];
  } else if (r < 24576) {
    int q = r - 8192; int j = q >> 12; q &= 4095; int n = q >> 6, k = q & 63;
    dst[r] = (_Float16)aw[((li * 4 + j) * 64 + k) * 64 + n];
  } else if (r < 32768) {
    int q = r - 24576; int n = q >> 6, k = q & 63;     // w1: K=64, N=128
    dst[r] = (_Float16)f1[li * 8192 + k * 128 + n];
  } else {
    int q = r - 32768; int n = q >> 7, k = q & 127;    // w2: K=128, N=64
    dst[r] = (_Float16)f2[li * 8192 + k * 64 + n];
  }
}

// ---------------------------------------------------------------------------
// Main fused forward: one block (128 thr / 4 waves) per batch element.
// ---------------------------------------------------------------------------
#define SMEM_BYTES 86016

__global__ void __launch_bounds__(128)
fsa_forward(const int* __restrict__ tok, const float* __restrict__ emb,
            const float* __restrict__ np_w, const float* __restrict__ np_b,
            const float* __restrict__ grp_bq, const float* __restrict__ grp_bk,
            const float* __restrict__ grp_ln_g, const float* __restrict__ grp_ln_b,
            const float* __restrict__ attn_b,
            const float* __restrict__ ln1_g, const float* __restrict__ ln1_b,
            const float* __restrict__ ln2_g, const float* __restrict__ ln2_b,
            const float* __restrict__ ffn_b1, const float* __restrict__ ffn_b2,
            const float* __restrict__ enc_g, const float* __restrict__ enc_b,
            const float* __restrict__ red_w, const float* __restrict__ red_b,
            const float* __restrict__ opc_w, const float* __restrict__ opc_b,
            const float* __restrict__ res_w, const float* __restrict__ res_b,
            const _Float16* __restrict__ wf, float* __restrict__ out) {
  extern __shared__ __align__(16) unsigned char smem[];
  float*    xs   = (float*)smem;                 // 64x64 x (persistent)
  float*    Pf   = (float*)(smem + 16384);       // qf / scores / y1(f16) / xn
  float*    Qf   = (float*)(smem + 32768);       // kf -> later qh16|kh16
  _Float16* a16  = (_Float16*)(smem + 49152);    // c / h / p / h2 staging
  _Float16* vT16 = (_Float16*)(smem + 57344);    // v transposed [d][token]
  _Float16* g16  = (_Float16*)(smem + 65536);    // group prob matrix
  _Float16* o16  = (_Float16*)(smem + 73728);    // attention output staging
  unsigned char* SB = smem + 81920;
  int*   ids    = (int*)(SB + 0);
  float* maskv  = (float*)(SB + 256);
  float* digv   = (float*)(SB + 512);
  float* isdg   = (float*)(SB + 768);
  float* pup    = (float*)(SB + 1024);           // later: left mask
  float* pdn    = (float*)(SB + 1280);           // later: right mask
  float* pdg    = (float*)(SB + 1536);           // later: op logits
  float* nbd    = (float*)(SB + 1792);           // later: scalars
  float* ltA    = (float*)(SB + 2048);
  float* Cp     = (float*)(SB + 2304);
  float* prup   = (float*)(SB + 2560);
  float* prdg   = (float*)(SB + 2816);
  float* sred   = (float*)(SB + 3072);
  float* rwv    = (float*)(SB + 3328);
  float* cumv   = (float*)(SB + 3584);
  float* pooled = (float*)(SB + 3840);

  const int b = blockIdx.x;
  const int tid = threadIdx.x;

  // ---- token features ----
  if (tid < 64) {
    int id = tok[b * 64 + tid];
    ids[tid] = id;
    maskv[tid] = (id != 0) ? 1.f : 0.f;
    float isd = (id >= 4 && id <= 13) ? 1.f : 0.f;
    isdg[tid] = isd;
    digv[tid] = ((float)id - 4.f) * isd;
  }
  __syncthreads();

  // ---- x = emb*sqrt(H) + numfeat@np_w + np_b + pe ----
  for (int idx = tid; idx < 4096; idx += 128) {
    int t = idx >> 6, h = idx & 63;
    int id = ids[t];
    float isop = (id >= 14 && id <= 17) ? 1.f : 0.f;
    float opty = isop * ((float)id - 13.f);
    float freq = __expf((float)(h & ~1) * (-9.2103403719761836f / 64.f));
    float ang = (float)t * freq;
    float pe = (h & 1) ? __cosf(ang) : __sinf(ang);
    xs[idx] = emb[id * 64 + h] * 8.f
            + digv[t] * np_w[h] + isdg[t] * np_w[64 + h]
            + opty * np_w[128 + h] + isop * np_w[192 + h]
            + np_b[h] + pe;
  }
  __syncthreads();

  for (int li = 0; li < 2; ++li) {
    const _Float16* W   = wf + li * 40960;
    const _Float16* wqt = W;
    const _Float16* wkt = W + 4096;
    const _Float16* awq = W + 8192;
    const _Float16* awk = W + 12288;
    const _Float16* awv = W + 16384;
    const _Float16* awo = W + 20480;
    const _Float16* w1t = W + 24576;
    const _Float16* w2t = W + 32768;
    const float* bq = grp_bq + li * 64;
    const float* bk = grp_bk + li * 64;
    const float* ab = attn_b + li * 256;
    const float* b1 = ffn_b1 + li * 128;
    const float* b2 = ffn_b2 + li * 64;

    // ======== Tree-Transformer group (constituent) attention ========
    layernorm16(xs, grp_ln_g + li * 64, grp_ln_b + li * 64, a16, tid);
    __syncthreads();
    float* qf = Pf; float* kf = Qf;
    wmma_gemm<64, 64>(a16, 64, wqt, 64, tid,
        [=](int r, int c, float v) { qf[r * 64 + c] = v + bq[c]; });
    wmma_gemm<64, 64>(a16, 64, wkt, 64, tid,
        [=](int r, int c, float v) { kf[r * 64 + c] = v + bk[c]; });
    __syncthreads();

    // tridiagonal softmax probs (only entries the math consumes)
    if (tid < 64) {
      int i = tid;
      bool vu = (i < 63) && maskv[i] > 0.f && maskv[i + 1] > 0.f;
      bool vd = (i > 0)  && maskv[i] > 0.f && maskv[i - 1] > 0.f;
      float su = -1e30f, sd = -1e30f;
      if (vu) { float d = 0.f; for (int j = 0; j < 64; ++j) d += qf[i*64+j]*kf[(i+1)*64+j]; su = d * 0.125f; }
      if (vd) { float d = 0.f; for (int j = 0; j < 64; ++j) d += qf[i*64+j]*kf[(i-1)*64+j]; sd = d * 0.125f; }
      float p_u, p_d, p_g;
      if (vu || vd) {
        float m = fmaxf(su, sd);
        float eu = vu ? __expf(su - m) : 0.f;
        float ed = vd ? __expf(sd - m) : 0.f;
        float rz = 1.f / (eu + ed);
        p_u = eu * rz; p_d = ed * rz; p_g = 0.f;
      } else { p_u = p_d = p_g = 0.015625f; }
      pup[i] = p_u; pdn[i] = p_d; pdg[i] = p_g;
    }
    __syncthreads();
    if (tid < 64) {
      int i = tid;
      float prd = (li == 0) ? 0.f : prdg[i];
      float nd = sqrtf(pdg[i] * pdg[i] + 1e-9f);
      nbd[i] = prd + (1.f - prd) * nd;
      if (i < 63) {
        float pru = (li == 0) ? 0.f : prup[i];
        float nb = sqrtf(pup[i] * pdn[i + 1] + 1e-9f);
        nb = pru + (1.f - pru) * nb;
        ltA[i] = __logf(nb + 1e-9f);
      }
    }
    __syncthreads();
    if (tid == 0) {
      Cp[0] = 0.f;
      for (int i = 0; i < 63; ++i) Cp[i + 1] = Cp[i] + ltA[i];
    }
    __syncthreads();
    // g[i][j] = exp(C[max]-C[min]) + 1e-9 offdiag, nbd on diag
    for (int idx = tid; idx < 4096; idx += 128) {
      int i = idx >> 6, j = idx & 63;
      float v;
      if (i == j) v = nbd[i];
      else { int hi = i > j ? i : j, lo = i > j ? j : i; v = __expf(Cp[hi] - Cp[lo]) + 1e-9f; }
      g16[idx] = (_Float16)v;
    }
    if (tid < 64) {  // save prior (only entries next layer reads)
      prdg[tid] = nbd[tid];
      if (tid < 63) prup[tid] = __expf(ltA[tid]) + 1e-9f;
    }
    __syncthreads();

    // ======== multi-head attention weighted by g ========
    layernorm16(xs, ln1_g + li * 64, ln1_b + li * 64, a16, tid);
    __syncthreads();
    _Float16* qh16 = (_Float16*)Qf;
    _Float16* kh16 = ((_Float16*)Qf) + 4096;
    wmma_gemm<64, 64>(a16, 64, awq, 64, tid,
        [=](int r, int c, float v) { qh16[r * 64 + c] = (_Float16)(v + ab[c]); });
    wmma_gemm<64, 64>(a16, 64, awk, 64, tid,
        [=](int r, int c, float v) { kh16[r * 64 + c] = (_Float16)(v + ab[64 + c]); });
    wmma_gemm<64, 64>(a16, 64, awv, 64, tid,
        [=](int r, int c, float v) { vT16[c * 64 + r] = (_Float16)(v + ab[128 + c]); });
    __syncthreads();

    _Float16* p16 = a16;   // h no longer needed
    float* sf = Pf;
    for (int hd = 0; hd < 4; ++hd) {
      // scores: qh_head @ kh_head^T / 4, key mask
      wmma_gemm<64, 16>(qh16 + hd * 16, 64, kh16 + hd * 16, 64, tid,
          [=](int r, int c, float v) {
            sf[r * 64 + c] = (maskv[c] > 0.f) ? v * 0.25f : -1e9f;
          });
      __syncthreads();
      if (tid < 64) {        // softmax row, times g
        int i = tid;
        float m = -1e30f;
        for (int j = 0; j < 64; ++j) m = fmaxf(m, sf[i * 64 + j]);
        float Z = 0.f;
        for (int j = 0; j < 64; ++j) Z += __expf(sf[i * 64 + j] - m);
        float rz = 1.f / Z;
        for (int j = 0; j < 64; ++j)
          p16[i * 64 + j] =
              (_Float16)(__expf(sf[i * 64 + j] - m) * rz * (float)g16[i * 64 + j]);
      }
      __syncthreads();
      // o_head = p @ v_head  (Bt = vT16 head slab, [d][token])
      wmma_gemm<16, 64>(p16, 64, vT16 + hd * 1024, 64, tid,
          [=](int r, int c, float v) { o16[r * 64 + hd * 16 + c] = (_Float16)v; });
      __syncthreads();
    }
    // output projection + residual
    wmma_gemm<64, 64>(o16, 64, awo, 64, tid,
        [=](int r, int c, float v) { xs[r * 64 + c] += v + ab[192 + c]; });
    __syncthreads();

    // ======== FFN ========
    layernorm16(xs, ln2_g + li * 64, ln2_b + li * 64, a16, tid);
    __syncthreads();
    _Float16* y1 = (_Float16*)Pf;
    wmma_gemm<128, 64>(a16, 64, w1t, 64, tid,
        [=](int r, int c, float v) {
          float t = v + b1[c];
          y1[r * 128 + c] = (_Float16)(t > 0.f ? t : 0.f);
        });
    __syncthreads();
    wmma_gemm<64, 128>(y1, 128, w2t, 128, tid,
        [=](int r, int c, float v) { xs[r * 64 + c] += v + b2[c]; });
    __syncthreads();
  }

  // ======== readout ========
  float* xn = Pf;
  layernorm32(xs, enc_g, enc_b, xn, tid);
  __syncthreads();
  if (tid < 64) {
    float d = red_b[0];
    for (int j = 0; j < 64; ++j) d += xn[tid * 64 + j] * red_w[j];
    sred[tid] = d + (1.f - maskv[tid]) * -1e9f;
  }
  __syncthreads();
  float* lmask = pup; float* rmask = pdn; float* oplog = pdg; float* scal = nbd;
  if (tid == 0) {
    float m = -1e30f;
    for (int i = 0; i < 64; ++i) m = fmaxf(m, sred[i]);
    float Z = 0.f;
    for (int i = 0; i < 64; ++i) { float e = __expf(sred[i] - m); rwv[i] = e; Z += e; }
    float rz = 1.f / Z, c = 0.f;
    for (int i = 0; i < 64; ++i) { rwv[i] *= rz; c += rwv[i]; cumv[i] = c; }
    for (int i = 0; i < 64; ++i) {
      lmask[i] = (1.f - cumv[i]) * isdg[i];
      rmask[i] = (cumv[i] - rwv[i]) * isdg[i];
    }
    scal[0] = soft_assemble(lmask, digv);
    scal[1] = soft_assemble(rmask, digv);
  }
  __syncthreads();
  if (tid < 64) {
    float p = 0.f;
    for (int t = 0; t < 64; ++t) p += xn[t * 64 + tid] * rwv[t];
    pooled[tid] = p;
  }
  __syncthreads();
  if (tid < 4) {
    float d = opc_b[tid];
    for (int j = 0; j < 64; ++j) d += pooled[j] * opc_w[j * 4 + tid];
    oplog[tid] = d;
  }
  __syncthreads();
  if (tid == 0) {
    int am = 0; float bm = oplog[0];
    for (int c = 1; c < 4; ++c) if (oplog[c] > bm) { bm = oplog[c]; am = c; }
    float l = scal[0], r = scal[1];
    bool bad = fabsf(r) < 1e-6f;
    float res, val = 1.f;
    if (am == 0) res = l + r;
    else if (am == 1) res = l - r;
    else if (am == 2) res = l * r;
    else { res = bad ? 0.f : l / r; val = bad ? 0.f : 1.f; }
    scal[2] = res; scal[3] = val;
    float sgn = (res > 0.f) ? 1.f : ((res < 0.f) ? -1.f : 0.f);
    scal[4] = sgn * log1pf(fabsf(res));
  }
  __syncthreads();

  // outputs: result | result_embedding | valid | left | right | op_logits | rw
  const long long B = 4096;
  const long long o0 = 0, o1 = B, o2 = o1 + B * 64, o3 = o2 + B,
                  o4 = o3 + B, o5 = o4 + B, o6 = o5 + B * 4;
  if (tid == 0) {
    out[o0 + b] = scal[2];
    out[o2 + b] = scal[3];
    out[o3 + b] = scal[0];
    out[o4 + b] = scal[1];
  }
  if (tid < 4) out[o5 + (long long)b * 4 + tid] = oplog[tid];
  if (tid < 64) {
    out[o1 + (long long)b * 64 + tid] =
        scal[4] * res_w[tid] + scal[3] * res_w[64 + tid] + res_b[tid];
    out[o6 + (long long)b * 64 + tid] = rwv[tid];
  }
}

// ---------------------------------------------------------------------------
extern "C" void kernel_launch(void* const* d_in, const int* in_sizes, int n_in,
                              void* d_out, int out_size, void* d_ws, size_t ws_size,
                              hipStream_t stream) {
  (void)in_sizes; (void)n_in; (void)out_size; (void)ws_size;
  const int*   tok     = (const int*)d_in[0];
  const float* emb     = (const float*)d_in[1];
  const float* np_w    = (const float*)d_in[2];
  const float* np_b    = (const float*)d_in[3];
  const float* grp_wq  = (const float*)d_in[4];
  const float* grp_bq  = (const float*)d_in[5];
  const float* grp_wk  = (const float*)d_in[6];
  const float* grp_bk  = (const float*)d_in[7];
  const float* grp_ln_g= (const float*)d_in[8];
  const float* grp_ln_b= (const float*)d_in[9];
  const float* attn_w  = (const float*)d_in[10];
  const float* attn_b  = (const float*)d_in[11];
  const float* ln1_g   = (const float*)d_in[12];
  const float* ln1_b   = (const float*)d_in[13];
  const float* ln2_g   = (const float*)d_in[14];
  const float* ln2_b   = (const float*)d_in[15];
  const float* ffn_w1  = (const float*)d_in[16];
  const float* ffn_b1  = (const float*)d_in[17];
  const float* ffn_w2  = (const float*)d_in[18];
  const float* ffn_b2  = (const float*)d_in[19];
  const float* enc_g   = (const float*)d_in[20];
  const float* enc_b   = (const float*)d_in[21];
  const float* red_w   = (const float*)d_in[22];
  const float* red_b   = (const float*)d_in[23];
  const float* opc_w   = (const float*)d_in[24];
  const float* opc_b   = (const float*)d_in[25];
  const float* res_w   = (const float*)d_in[26];
  const float* res_b   = (const float*)d_in[27];

  _Float16* wf = (_Float16*)d_ws;   // 2*40960 halves = 160 KB packed f16 weights

  convert_weights<<<(2 * 40960 + 255) / 256, 256, 0, stream>>>(
      grp_wq, grp_wk, attn_w, ffn_w1, ffn_w2, wf);

  fsa_forward<<<4096, 128, SMEM_BYTES, stream>>>(
      tok, emb, np_w, np_b, grp_bq, grp_bk, grp_ln_g, grp_ln_b, attn_b,
      ln1_g, ln1_b, ln2_g, ln2_b, ffn_b1, ffn_b2, enc_g, enc_b,
      red_w, red_b, opc_w, opc_b, res_w, res_b, wf, (float*)d_out);
}
